// MultiHeadAttention_49237505081571
// MI455X (gfx1250) — matrix operand order repaired and checked by reading, not compiled, not run on hardware
//
#include <hip/hip_runtime.h>
#include <hip/hip_bf16.h>

typedef __attribute__((ext_vector_type(16))) _Float16 v16h;
typedef __attribute__((ext_vector_type(8)))  _Float16 v8h;
typedef __attribute__((ext_vector_type(8)))  float    v8f;

#define CB 2
#define CS 2048
#define CD 1024
#define CH 16
#define CHD 64

__device__ __forceinline__ v16h cat8(v8h lo, v8h hi) {
    v16h r;
#pragma unroll
    for (int i = 0; i < 8; ++i) { r[i] = lo[i]; r[i + 8] = hi[i]; }
    return r;
}

// A-fragment (16x32 f16, M x K): lane m in [0,15] holds row m, K = {0..7, 16..23};
// lane m+16 holds row m, K = {8..15, 24..31}.   src row-major, stride in halves.
__device__ __forceinline__ v16h load_a_frag(const _Float16* base, int stride, int lane) {
    const int m  = lane & 15;
    const int kh = (lane >> 4) << 3;          // 0 or 8
    const _Float16* p = base + (size_t)m * stride + kh;
    v8h lo = *(const v8h*)(p);
    v8h hi = *(const v8h*)(p + 16);
    return cat8(lo, hi);
}

// B-fragment (32x16 f16, K x N): lane n in [0,15] holds column n, K = {0..7, 16..23};
// lane n+16 holds column n, K = {8..15, 24..31}.   src row-major, stride in halves.
__device__ __forceinline__ v16h load_b_frag(const _Float16* base, int stride, int lane) {
    const int n  = lane & 15;
    const int kh = (lane >> 4) << 3;          // 0 or 8
    v16h r;
#pragma unroll
    for (int i = 0; i < 8; ++i) {
        r[i]     = base[(size_t)(kh + i) * stride + n];
        r[i + 8] = base[(size_t)(kh + 16 + i) * stride + n];
    }
    return r;
}

__device__ __forceinline__ v8f wmma_f16(v16h a, v16h b, v8f c) {
    return __builtin_amdgcn_wmma_f32_16x16x32_f16(false, a, false, b, (short)0, c, false, false);
}

// ---------------- conversion kernels ----------------

__global__ void cvt_x_kernel(const float* __restrict__ x, _Float16* __restrict__ xh, int n) {
    int i = blockIdx.x * blockDim.x + threadIdx.x;
    if (i < n) xh[i] = (_Float16)x[i];
}

// Wt[k][n] = W[n][k]  (f16), D x D
__global__ void cvt_wt_kernel(const float* __restrict__ w, _Float16* __restrict__ wt) {
    int o = blockIdx.x * blockDim.x + threadIdx.x;   // output-coalesced
    int k = o >> 10, n = o & (CD - 1);
    wt[o] = (_Float16)w[(size_t)n * CD + k];
}

// ---------------- QKV projection GEMM ----------------
// Y[m][n] = sum_k X[m][k] * W[n][k] + b[n] = X @ Wt (+ b)
// One wave -> 32(m) x 64(n) tile. Ping-pong double-buffered K-loop (unroll x2):
// fragments for the next 32-K step are in flight while the 8 WMMAs of the
// current step execute, with no register-rotation copies.
// mode: 0=Q (scale 1/8, [b,h,s,d]), 1=K ([b,h,d,s]), 2=V ([b,h,s,d])
__global__ void __launch_bounds__(256)
gemm_qkv_kernel(const _Float16* __restrict__ Xh, const _Float16* __restrict__ Wt,
                const float* __restrict__ bias, _Float16* __restrict__ out, int mode) {
    const int lane = threadIdx.x & 31;
    const int wid  = threadIdx.x >> 5;
    const int m0 = (blockIdx.x * 8 + wid) * 32;
    const int n0 = blockIdx.y * 64;

    v8f acc[2][4] = {{{}, {}, {}, {}}, {{}, {}, {}, {}}};

    v16h a0[2], a1[2], bf[2][4];

    auto loadset = [&](int kk, int p) {
        a0[p] = load_a_frag(Xh + (size_t)m0 * CD + kk,        CD, lane);
        a1[p] = load_a_frag(Xh + (size_t)(m0 + 16) * CD + kk, CD, lane);
#pragma unroll
        for (int t = 0; t < 4; ++t)
            bf[p][t] = load_b_frag(Wt + (size_t)kk * CD + n0 + t * 16, CD, lane);
    };
    auto compute = [&](int p) {
#pragma unroll
        for (int t = 0; t < 4; ++t) {
            acc[0][t] = wmma_f16(a0[p], bf[p][t], acc[0][t]);
            acc[1][t] = wmma_f16(a1[p], bf[p][t], acc[1][t]);
        }
    };

    loadset(0, 0);
    for (int kk = 0; kk < CD; kk += 64) {
        loadset(kk + 32, 1);       // kk+32 <= CD-32 always (CD % 64 == 0)
        compute(0);
        if (kk + 64 < CD) loadset(kk + 64, 0);
        compute(1);
    }

    const int rowoff = (lane >> 4) << 3;     // 0 or 8
    const int colo   = lane & 15;
#pragma unroll
    for (int u = 0; u < 2; ++u) {
#pragma unroll
        for (int t = 0; t < 4; ++t) {
            const int n = n0 + t * 16 + colo;
            const float bv = bias[n];
            const int h = n >> 6, d = n & 63;
#pragma unroll
            for (int r = 0; r < 8; ++r) {
                const int row = m0 + u * 16 + r + rowoff;
                const int bb = row >> 11, s = row & (CS - 1);
                float v = acc[u][t][r] + bv;
                if (mode == 0) {
                    out[((((size_t)bb * CH + h) * CS) + s) * CHD + d] = (_Float16)(v * 0.125f);
                } else if (mode == 1) {
                    out[((((size_t)bb * CH + h) * CHD) + d) * CS + s] = (_Float16)v;
                } else {
                    out[((((size_t)bb * CH + h) * CS) + s) * CHD + d] = (_Float16)v;
                }
            }
        }
    }
}

// ---------------- flash attention ----------------
// Qh: [B,H,S,HD] f16 (pre-scaled by 1/8), Kt: [B,H,HD,S] f16, Vh: [B,H,S,HD] f16
// One wave handles a 16-row Q tile; streams keys 32 at a time.
__global__ void __launch_bounds__(256)
attn_kernel(const _Float16* __restrict__ Qh, const _Float16* __restrict__ Kt,
            const _Float16* __restrict__ Vh, const float* __restrict__ mask,
            float* __restrict__ out) {
    const int lane = threadIdx.x & 31;
    const int wid  = threadIdx.x >> 5;
    const int b = blockIdx.z, h = blockIdx.y;
    const int q0 = blockIdx.x * 128 + wid * 16;
    const size_t bh = (size_t)b * CH + h;

    const _Float16* Qb  = Qh + (bh * CS + q0) * CHD;
    const _Float16* Ktb = Kt + bh * CHD * CS;
    const _Float16* Vb  = Vh + bh * CS * CHD;
    const float*    mb  = mask + (size_t)b * CS;

    const v16h qa0 = load_a_frag(Qb,      CHD, lane);   // d = 0..31
    const v16h qa1 = load_a_frag(Qb + 32, CHD, lane);   // d = 32..63

    v8f o[4] = {{}, {}, {}, {}};
    float rmax[8], rsum[8];
#pragma unroll
    for (int r = 0; r < 8; ++r) { rmax[r] = -1e30f; rsum[r] = 0.0f; }

    __shared__ __align__(16) _Float16 pbuf[8 * 512];
    _Float16* pb = pbuf + wid * 512;

    const int rowoff = (lane >> 4) << 3;
    const int colo   = lane & 15;

    for (int s0 = 0; s0 < CS; s0 += 32) {
        // scores: [16q x 32k] via two 16x16 tiles, each accumulated over d in 2 steps
        v8f sc0 = {}, sc1 = {};
        {
            v16h kb = load_b_frag(Ktb + s0,            CS, lane);         // d rows 0..31,  keys s0..s0+15
            sc0 = wmma_f16(qa0, kb, sc0);
            kb = load_b_frag(Ktb + 32 * CS + s0,       CS, lane);         // d rows 32..63
            sc0 = wmma_f16(qa1, kb, sc0);
            kb = load_b_frag(Ktb + s0 + 16,            CS, lane);         // keys s0+16..s0+31
            sc1 = wmma_f16(qa0, kb, sc1);
            kb = load_b_frag(Ktb + 32 * CS + s0 + 16,  CS, lane);
            sc1 = wmma_f16(qa1, kb, sc1);
        }
        const float mv0 = mb[s0 + colo];
        const float mv1 = mb[s0 + 16 + colo];

        float scale[8];
#pragma unroll
        for (int r = 0; r < 8; ++r) {
            sc0[r] += mv0;
            sc1[r] += mv1;
            float v = fmaxf(sc0[r], sc1[r]);
            v = fmaxf(v, __shfl_xor(v, 1, 32));
            v = fmaxf(v, __shfl_xor(v, 2, 32));
            v = fmaxf(v, __shfl_xor(v, 4, 32));
            v = fmaxf(v, __shfl_xor(v, 8, 32));
            const float nm = fmaxf(rmax[r], v);
            scale[r] = __expf(rmax[r] - nm);
            rmax[r] = nm;
            sc0[r] = __expf(sc0[r] - nm);
            sc1[r] = __expf(sc1[r] - nm);
            float sm = sc0[r] + sc1[r];
            sm += __shfl_xor(sm, 1, 32);
            sm += __shfl_xor(sm, 2, 32);
            sm += __shfl_xor(sm, 4, 32);
            sm += __shfl_xor(sm, 8, 32);
            rsum[r] = rsum[r] * scale[r] + sm;
        }
#pragma unroll
        for (int t = 0; t < 4; ++t)
#pragma unroll
            for (int r = 0; r < 8; ++r) o[t][r] *= scale[r];

        // bounce P (C-layout) through LDS into A-fragment layout
#pragma unroll
        for (int r = 0; r < 8; ++r) {
            pb[(r + rowoff) * 32 + colo]      = (_Float16)sc0[r];
            pb[(r + rowoff) * 32 + 16 + colo] = (_Float16)sc1[r];
        }
        __syncthreads();
        v16h pa;
        {
            const int m  = lane & 15;
            const int kh = (lane >> 4) << 3;
            v8h l0 = *(const v8h*)(pb + m * 32 + kh);
            v8h l1 = *(const v8h*)(pb + m * 32 + kh + 16);
            pa = cat8(l0, l1);
        }
        __syncthreads();

        // O += P @ V   (B-fragment: lane l holds V column t*16 + l%16, 16 keys strided by CHD)
        const _Float16* vb = Vb + (size_t)s0 * CHD;
#pragma unroll
        for (int t = 0; t < 4; ++t) {
            v16h vf = load_b_frag(vb + t * 16, CHD, lane);
            o[t] = wmma_f16(pa, vf, o[t]);
        }
    }

    // epilogue: normalize and store fp32 [B,S,D]
#pragma unroll
    for (int t = 0; t < 4; ++t) {
#pragma unroll
        for (int r = 0; r < 8; ++r) {
            const int row = q0 + r + rowoff;
            const int col = h * CHD + t * 16 + colo;
            out[((size_t)b * CS + row) * CD + col] = o[t][r] / rsum[r];
        }
    }
}

// ---------------- host launcher ----------------

extern "C" void kernel_launch(void* const* d_in, const int* in_sizes, int n_in,
                              void* d_out, int out_size, void* d_ws, size_t ws_size,
                              hipStream_t stream) {
    (void)in_sizes; (void)n_in; (void)out_size; (void)ws_size;

    const float* hs   = (const float*)d_in[0];
    const float* mask = (const float*)d_in[1];
    const float* Wq   = (const float*)d_in[2];
    const float* bq   = (const float*)d_in[3];
    const float* Wk   = (const float*)d_in[4];
    const float* bk   = (const float*)d_in[5];
    const float* Wv   = (const float*)d_in[6];
    const float* bv   = (const float*)d_in[7];
    float* out = (float*)d_out;

    char* ws = (char*)d_ws;
    const size_t xh_bytes = (size_t)CB * CS * CD * 2;       // 8 MB
    const size_t wt_bytes = (size_t)CD * CD * 2;            // 2 MB each
    const size_t pr_bytes = (size_t)CB * CH * CS * CHD * 2; // 8 MB each

    _Float16* Xh  = (_Float16*)ws;
    _Float16* Wtq = (_Float16*)(ws + xh_bytes);
    _Float16* Wtk = (_Float16*)(ws + xh_bytes + wt_bytes);
    _Float16* Wtv = (_Float16*)(ws + xh_bytes + 2 * wt_bytes);
    _Float16* Qh  = (_Float16*)(ws + xh_bytes + 3 * wt_bytes);
    _Float16* Kt  = (_Float16*)(ws + xh_bytes + 3 * wt_bytes + pr_bytes);
    _Float16* Vh  = (_Float16*)(ws + xh_bytes + 3 * wt_bytes + 2 * pr_bytes);

    const int nx = CB * CS * CD;
    cvt_x_kernel<<<nx / 256, 256, 0, stream>>>(hs, Xh, nx);
    cvt_wt_kernel<<<(CD * CD) / 256, 256, 0, stream>>>(Wq, Wtq);
    cvt_wt_kernel<<<(CD * CD) / 256, 256, 0, stream>>>(Wk, Wtk);
    cvt_wt_kernel<<<(CD * CD) / 256, 256, 0, stream>>>(Wv, Wtv);

    dim3 ggrid(16, 16);   // (M/256, N/64): M = B*S = 4096, N = D = 1024
    gemm_qkv_kernel<<<ggrid, 256, 0, stream>>>(Xh, Wtq, bq, Qh, 0);
    gemm_qkv_kernel<<<ggrid, 256, 0, stream>>>(Xh, Wtk, bk, Kt, 1);
    gemm_qkv_kernel<<<ggrid, 256, 0, stream>>>(Xh, Wtv, bv, Vh, 2);

    dim3 agrid(CS / 128, CH, CB);
    attn_kernel<<<agrid, 256, 0, stream>>>(Qh, Kt, Vh, mask, out);
}
